// TemporallyAdaptiveLinear_42039139893868
// MI455X (gfx1250) — matrix operand order repaired and checked by reading, not compiled, not run on hardware
//
#include <hip/hip_runtime.h>
#include <stdint.h>

// ---------------- problem constants (from reference setup_inputs) ----------
#define BDIM 16
#define TDIM 8
#define NTOK 197
#define CDIM 768
#define GRP  3
#define BT   (BDIM * TDIM)   // 128
#define OC   (GRP * CDIM)    // 2304

// ---------------- GEMM tiling ----------------------------------------------
#define MT   128             // output-channel tile per block
#define NT   64              // token tile per block
#define KB   32              // K chunk (== WMMA K for bf16)
#define LSTR 40              // LDS row stride in bf16 elems (32 data + 8 pad -> 80B, bank-spread)

typedef __attribute__((ext_vector_type(16))) __bf16 v16bf;
typedef __attribute__((ext_vector_type(2)))  __bf16 v2bf;
typedef __attribute__((ext_vector_type(8)))  float  v8f;
typedef __attribute__((ext_vector_type(2)))  float  v2f;

union FragBF { v16bf v; uint4 q[2]; };

// f32x2 -> packed bf16x2 (lowers to v_cvt_pk_bf16_f32, RTNE)
static __device__ __forceinline__ unsigned int packpair(float lo, float hi) {
  v2f f; f.x = lo; f.y = hi;
  v2bf b = __builtin_convertvector(f, v2bf);
  union { v2bf b; unsigned int u; } cvt; cvt.b = b;
  return cvt.u;
}

static __device__ __forceinline__ uint4 pack8(float a0, float a1, float a2, float a3,
                                              float b0, float b1, float b2, float b3) {
  uint4 r;
  r.x = packpair(a0, a1);
  r.y = packpair(a2, a3);
  r.z = packpair(b0, b1);
  r.w = packpair(b2, b3);
  return r;
}

// ---------------------------------------------------------------------------
// Kernel 0: weight f32 -> bf16 (one-time; L2-resident afterwards)
// ---------------------------------------------------------------------------
__global__ __launch_bounds__(256) void cvt_w_kernel(const float* __restrict__ w,
                                                    unsigned short* __restrict__ wbf) {
  int i = (blockIdx.x * 256 + threadIdx.x) * 4;
  if (i < OC * CDIM) {
    float4 v = *(const float4*)(w + i);
    uint2 p;
    p.x = packpair(v.x, v.y);
    p.y = packpair(v.z, v.w);
    *(uint2*)(wbf + i) = p;
  }
}

// ---------------------------------------------------------------------------
// Kernel 1: routing scales
// s[bt][oc] = 1 + a_bias[oc] + sum_{ic,k} cls[b, ic, t+k-1] * a_weight[oc, ic, k]
// ---------------------------------------------------------------------------
__global__ __launch_bounds__(256) void routing_kernel(const float* __restrict__ x,
                                                      const float* __restrict__ aw,
                                                      const float* __restrict__ ab,
                                                      float* __restrict__ sarr) {
  const int oc = blockIdx.x * 256 + threadIdx.x;   // 0..2303
  const int t  = blockIdx.y;                       // 0..7
  const int b  = blockIdx.z;                       // 0..15

  const float* wrow = aw + (size_t)oc * (CDIM * 3);
  const float* c0 = x + (size_t)(b * TDIM + t) * NTOK * CDIM;   // CLS token row
  const float* cm = (t > 0)        ? (c0 - (size_t)NTOK * CDIM) : nullptr;
  const float* cp = (t < TDIM - 1) ? (c0 + (size_t)NTOK * CDIM) : nullptr;

  float acc = ab[oc] + 1.0f;
  #pragma unroll 8
  for (int ic = 0; ic < CDIM; ++ic) {
    const float w0 = wrow[3 * ic + 0];
    const float w1 = wrow[3 * ic + 1];
    const float w2 = wrow[3 * ic + 2];
    if (cm) acc = fmaf(cm[ic], w0, acc);
    acc = fmaf(c0[ic], w1, acc);
    if (cp) acc = fmaf(cp[ic], w2, acc);
  }
  sarr[(size_t)(b * TDIM + t) * OC + oc] = acc;
}

// ---------------------------------------------------------------------------
// Kernel 2: batched modulated GEMM with BF16 WMMA, double-buffered LDS,
// K-loop hand-unrolled by 2 so LDS buffer offsets are literals.
// Per (bt,g):  Out[o,n] = sum_i W[g,o,i] * s[bt,g,i] * x[bt,n,i] + bias[g,o]*s[bt,g,o]
// A tile (bf16 weights) staged global->LDS with global_load_async_to_lds_b128
// (ASYNCcnt), B tile scaled+converted through VGPRs. One barrier / K-chunk.
// ---------------------------------------------------------------------------
template <bool WB>
__global__ __launch_bounds__(256) void tal_gemm_kernel(
    const float* __restrict__ x,
    const float* __restrict__ wf32,
    const unsigned short* __restrict__ wbf,
    const float* __restrict__ sarr,
    const float* __restrict__ bias,
    float* __restrict__ out) {
  __shared__ alignas(16) unsigned short As[2][MT * LSTR];
  __shared__ alignas(16) unsigned short Bs[2][NT * LSTR];
  __shared__ float bmod[MT];

  const int batch = blockIdx.x;         // 0..383
  const int bt = batch / GRP;
  const int g  = batch % GRP;
  const int o_base = blockIdx.y * MT;   // 0,128,...,640
  const int n0 = blockIdx.z * NT;       // 0,64,128,192
  const int tid = threadIdx.x;

  const float* sb = sarr + (size_t)bt * OC + g * CDIM;   // scales for this (bt,g)
  if (tid < MT)
    bmod[tid] = bias[g * CDIM + o_base + tid] * sb[o_base + tid];

  const int lane   = tid & 31;
  const int wv     = tid >> 5;
  const int wm     = wv >> 1;             // 0..3 -> M offset wm*32
  const int wn     = wv & 1;              // 0..1 -> N offset wn*32
  const int colx   = lane & 15;           // C/D: N column within 16-tile
  const int rowoff = (lane >> 4) << 3;    // C/D: +8 rows for upper half-wave
  const int k0a = (lane < 16) ? 0 : 8;    // ISA 16-bit A-matrix lane K swizzle
  const int k0b = (lane < 16) ? 0 : 8;    // B-matrix lane K swizzle (matches A)

  const float* xrow = x + (size_t)bt * NTOK * CDIM;

  // staging coordinates
  const int am  = tid >> 1;               // 0..127 : A row (o)
  const int akq = (tid & 1) << 4;         // 0/16   : A k-half (2x16B, contiguous)
  const int bn  = tid >> 2;               // 0..63  : B row (n)
  const int bkq = (tid & 3) << 3;         // 0,8,16,24
  const int gn  = n0 + bn;
  const float* xpB = (gn < NTOK) ? (xrow + (size_t)gn * CDIM + bkq) : nullptr;
  const float* spB = sb + bkq;

  // per-thread A source / LDS destination for the async DMA path
  const unsigned short* wpA =
      wbf + (size_t)(g * CDIM + o_base + am) * CDIM + akq;
  const float* wpAf =
      wf32 + (size_t)(g * CDIM + o_base + am) * CDIM + akq;
  const unsigned ldsA0 = (unsigned)(uintptr_t)&As[0][am * LSTR + akq];
  const unsigned ldsAbuf = (unsigned)(MT * LSTR * sizeof(unsigned short));

  // lane-fixed fragment sources (byte-constant buffer offsets at use sites)
  const int mrow = wm * 32 + (lane & 15);
  const int ncol = wn * 32 + (lane & 15);
  const unsigned short* apL = &As[0][mrow * LSTR + k0a];   // +16 rows -> +16*LSTR
  const unsigned short* bpL = &Bs[0][ncol * LSTR + k0b];

  v8f acc00 = {}, acc01 = {}, acc10 = {}, acc11 = {};

  // ---- staging helpers -----------------------------------------------------
  auto stageA = [&](int kc, int buf) {
    if (WB) {
      // 32 bytes of bf16 weights per lane, DMA'd straight into LDS (ASYNCcnt)
      const unsigned long long ga = (unsigned long long)(uintptr_t)(wpA + kc);
      const unsigned la = ldsA0 + (unsigned)buf * ldsAbuf;
      asm volatile("global_load_async_to_lds_b128 %0, %1, off offset:0"
                   :: "v"(la), "v"(ga) : "memory");
      asm volatile("global_load_async_to_lds_b128 %0, %1, off offset:16"
                   :: "v"(la), "v"(ga) : "memory");
    } else {
      const float* wp = wpAf + kc;
      float4 a0 = *(const float4*)wp;
      float4 a1 = *(const float4*)(wp + 4);
      float4 a2 = *(const float4*)(wp + 8);
      float4 a3 = *(const float4*)(wp + 12);
      *(uint4*)&As[buf][am * LSTR + akq]     = pack8(a0.x, a0.y, a0.z, a0.w,
                                                     a1.x, a1.y, a1.z, a1.w);
      *(uint4*)&As[buf][am * LSTR + akq + 8] = pack8(a2.x, a2.y, a2.z, a2.w,
                                                     a3.x, a3.y, a3.z, a3.w);
    }
  };
  auto stageB = [&](int kc, int buf) {
    uint4 bpack;
    if (xpB) {
      float4 x0 = *(const float4*)(xpB + kc);
      float4 x1 = *(const float4*)(xpB + kc + 4);
      float4 s0 = *(const float4*)(spB + kc);
      float4 s1 = *(const float4*)(spB + kc + 4);
      bpack = pack8(x0.x * s0.x, x0.y * s0.y, x0.z * s0.z, x0.w * s0.w,
                    x1.x * s1.x, x1.y * s1.y, x1.z * s1.z, x1.w * s1.w);
      __builtin_prefetch(xpB + kc + KB, 0, 3);   // near prefetch of next chunk
    } else {
      bpack = make_uint4(0u, 0u, 0u, 0u);
    }
    *(uint4*)&Bs[buf][bn * LSTR + bkq] = bpack;
  };
  // compute one K-chunk from buffer `buf` (literal at every call site)
  auto compute = [&](int buf) {
    FragBF a0, a1, b0, b1;
    const unsigned short* ap = apL + buf * (MT * LSTR);
    a0.q[0] = *(const uint4*)ap;
    a0.q[1] = *(const uint4*)(ap + 16);
    a1.q[0] = *(const uint4*)(ap + 16 * LSTR);
    a1.q[1] = *(const uint4*)(ap + 16 * LSTR + 16);
    const unsigned short* bp = bpL + buf * (NT * LSTR);
    b0.q[0] = *(const uint4*)bp;
    b0.q[1] = *(const uint4*)(bp + 16);
    b1.q[0] = *(const uint4*)(bp + 16 * LSTR);
    b1.q[1] = *(const uint4*)(bp + 16 * LSTR + 16);

    acc00 = __builtin_amdgcn_wmma_f32_16x16x32_bf16(false, a0.v, false, b0.v,
                                                    (short)0, acc00, false, false);
    acc01 = __builtin_amdgcn_wmma_f32_16x16x32_bf16(false, a0.v, false, b1.v,
                                                    (short)0, acc01, false, false);
    acc10 = __builtin_amdgcn_wmma_f32_16x16x32_bf16(false, a1.v, false, b0.v,
                                                    (short)0, acc10, false, false);
    acc11 = __builtin_amdgcn_wmma_f32_16x16x32_bf16(false, a1.v, false, b1.v,
                                                    (short)0, acc11, false, false);
  };

  // ---- prologue: fill buffer 0 --------------------------------------------
  stageA(0, 0);
  stageB(0, 0);
  if (WB) asm volatile("s_wait_asynccnt 0x0" ::: "memory");
  __syncthreads();

  // ---- main loop: pairs of K-chunks; buffer index is a literal -------------
  // CDIM/KB = 24 chunks -> 12 pairs; within a pair, chunk kc+KB always exists.
  for (int kc = 0; kc < CDIM; kc += 2 * KB) {
    // chunk kc (buffer 0); stage kc+KB into buffer 1
    stageA(kc + KB, 1);
    stageB(kc + KB, 1);
    compute(0);
    if (WB) asm volatile("s_wait_asynccnt 0x0" ::: "memory");
    __syncthreads();

    // chunk kc+KB (buffer 1); stage kc+2*KB into buffer 0 (if any)
    const bool more = (kc + 2 * KB) < CDIM;
    if (more) {
      stageA(kc + 2 * KB, 0);
      stageB(kc + 2 * KB, 0);
    }
    compute(1);
    if (WB && more) asm volatile("s_wait_asynccnt 0x0" ::: "memory");
    __syncthreads();
  }

  // ---- epilogue: add modulated bias, store (out layout: [bt][3c][n]) -------
  const size_t obase = (size_t)bt * OC * NTOK;
  #pragma unroll
  for (int ti = 0; ti < 2; ++ti) {
    #pragma unroll
    for (int tj = 0; tj < 2; ++tj) {
      const v8f acc = (ti == 0) ? ((tj == 0) ? acc00 : acc01)
                                : ((tj == 0) ? acc10 : acc11);
      const int n = n0 + wn * 32 + tj * 16 + colx;
      if (n < NTOK) {
        const int mloc = wm * 32 + ti * 16 + rowoff;
        float* op = out + obase + (size_t)(g * CDIM + o_base + mloc) * NTOK + n;
        #pragma unroll
        for (int r = 0; r < 8; ++r)
          op[(size_t)r * NTOK] = acc[r] + bmod[mloc + r];
      }
    }
  }
}

// ---------------------------------------------------------------------------
extern "C" void kernel_launch(void* const* d_in, const int* in_sizes, int n_in,
                              void* d_out, int out_size, void* d_ws, size_t ws_size,
                              hipStream_t stream) {
  const float* x    = (const float*)d_in[0];   // (128, 197, 768)
  const float* aw   = (const float*)d_in[1];   // (2304, 768, 3)
  const float* ab   = (const float*)d_in[2];   // (2304,)
  const float* wt   = (const float*)d_in[3];   // (2304, 768)
  const float* bias = (const float*)d_in[4];   // (2304,)
  float* out = (float*)d_out;                  // (128, 2304, 197)

  float* sarr = (float*)d_ws;                                   // 128*2304 f32
  const size_t s_bytes = (size_t)BT * OC * sizeof(float);
  unsigned short* wbf = (unsigned short*)((char*)d_ws + s_bytes);
  const size_t need_wbf = s_bytes + (size_t)OC * CDIM * sizeof(unsigned short);
  const bool use_wbf = (ws_size >= need_wbf);

  routing_kernel<<<dim3(OC / 256, TDIM, BDIM), 256, 0, stream>>>(x, aw, ab, sarr);

  const dim3 ggrid(BT * GRP, CDIM / MT, (NTOK + NT - 1) / NT);  // 384 x 6 x 4
  if (use_wbf) {
    cvt_w_kernel<<<(OC * CDIM / 4 + 255) / 256, 256, 0, stream>>>(wt, wbf);
    tal_gemm_kernel<true><<<ggrid, 256, 0, stream>>>(x, wt, wbf, sarr, bias, out);
  } else {
    tal_gemm_kernel<false><<<ggrid, 256, 0, stream>>>(x, wt, nullptr, sarr, bias, out);
  }
}
